// DeepSeekV2MoE_40750649704539
// MI455X (gfx1250) — matrix operand order repaired and checked by reading, not compiled, not run on hardware
//
#include <hip/hip_runtime.h>
#include <math.h>

typedef __attribute__((ext_vector_type(16))) _Float16 v16h;
typedef __attribute__((ext_vector_type(8)))  _Float16 v8h;
typedef __attribute__((ext_vector_type(8)))  float    v8f;

#define H_ 2048
#define I_ 1408
#define E_ 8
#define T_ 512

// K-chunk staged in LDS, padded row stride (halfs)
#define KC_   512
#define LROW_ (KC_ + 8)

// ---- workspace layout (bytes) ----
#define OFF_CNT   0u
#define OFF_LIST  256u                                    // E*T ints  = 16384
#define OFF_WL    (OFF_LIST + E_*T_*4u)                   // E*T floats= 16384
#define OFF_XH    (OFF_WL + E_*T_*4u)                     // T*H halfs = 2 MB   (33024, 256-aligned)
#define OFF_ACT   (OFF_XH + (size_t)T_*H_*2u)             // E*T*I halfs = 11.5 MB
#define OFF_OUTK  (OFF_ACT + (size_t)E_*T_*I_*2u)         // T*2*H floats = 8 MB

// ------------------------------------------------------------------
// Kernel 1: fp32 -> f16 convert of x, plus zero expert counters
// ------------------------------------------------------------------
__global__ __launch_bounds__(256) void moe_convert(const float* __restrict__ x,
                                                   _Float16* __restrict__ xh,
                                                   int* __restrict__ cnt) {
    int tid = threadIdx.x;
    if (blockIdx.x == 0 && tid < E_) cnt[tid] = 0;
    size_t base = ((size_t)blockIdx.x * 256 + tid) * 8;   // 512 blocks * 256 * 8 = T*H
    float4 f0 = *(const float4*)(x + base);
    float4 f1 = *(const float4*)(x + base + 4);
    v8h h;
    h[0] = (_Float16)f0.x; h[1] = (_Float16)f0.y; h[2] = (_Float16)f0.z; h[3] = (_Float16)f0.w;
    h[4] = (_Float16)f1.x; h[5] = (_Float16)f1.y; h[6] = (_Float16)f1.z; h[7] = (_Float16)f1.w;
    *(v8h*)(xh + base) = h;
}

// ------------------------------------------------------------------
// Kernel 2: router — one wave32 per token
// ------------------------------------------------------------------
__global__ __launch_bounds__(256) void moe_router(const float* __restrict__ x,
                                                  const float* __restrict__ wg,
                                                  int* __restrict__ cnt,
                                                  int* __restrict__ list,
                                                  float* __restrict__ wl) {
    int lane = threadIdx.x & 31;
    int wave = threadIdx.x >> 5;
    int t = blockIdx.x * 8 + wave;

    float acc[E_];
#pragma unroll
    for (int e = 0; e < E_; ++e) acc[e] = 0.f;

    const float* xr = x + (size_t)t * H_;
    for (int h = lane; h < H_; h += 32) {
        float xv = xr[h];
#pragma unroll
        for (int e = 0; e < E_; ++e) acc[e] += xv * wg[e * H_ + h];
    }
#pragma unroll
    for (int off = 16; off; off >>= 1) {
#pragma unroll
        for (int e = 0; e < E_; ++e) acc[e] += __shfl_down(acc[e], off, 32);
    }
    if (lane == 0) {
        float m = acc[0];
#pragma unroll
        for (int e = 1; e < E_; ++e) m = fmaxf(m, acc[e]);
        float p[E_];
#pragma unroll
        for (int e = 0; e < E_; ++e) p[e] = __expf(acc[e] - m);
        // top-1 / top-2 (first-occurrence tie-break like lax.top_k)
        int i1 = 0; float s1 = p[0];
#pragma unroll
        for (int e = 1; e < E_; ++e) if (p[e] > s1) { s1 = p[e]; i1 = e; }
        int i2 = (i1 == 0) ? 1 : 0; float s2 = p[i2];
#pragma unroll
        for (int e = 0; e < E_; ++e) if (e != i1 && p[e] > s2) { s2 = p[e]; i2 = e; }
        float inv = 1.f / (s1 + s2);
        int pos1 = atomicAdd(&cnt[i1], 1);
        list[i1 * T_ + pos1] = t * 2 + 0;
        wl  [i1 * T_ + pos1] = s1 * inv;
        int pos2 = atomicAdd(&cnt[i2], 1);
        list[i2 * T_ + pos2] = t * 2 + 1;
        wl  [i2 * T_ + pos2] = s2 * inv;
    }
}

// ------------------------------------------------------------------
// Kernel 3: GEMM1 + SwiGLU.  Block = (expert, 32 tokens, 128 gate cols).
// A (16x32 f16) from LDS, B (32x16 f16) converted in-register from fp32 w1.
// ------------------------------------------------------------------
__global__ __launch_bounds__(256) void moe_gemm1(const float* __restrict__ w1,
                                                 const _Float16* __restrict__ xh,
                                                 const int* __restrict__ cnt,
                                                 const int* __restrict__ list,
                                                 _Float16* __restrict__ act) {
    constexpr int MT = T_ / 32;    // 16
    constexpr int NT = I_ / 128;   // 11
    int bx = blockIdx.x;
    int e  = bx / (MT * NT);
    int rm = bx % (MT * NT);
    int mt = rm / NT;
    int nt = rm % NT;
    int cnt_e = cnt[e];
    if (mt * 32 >= cnt_e) return;

    __shared__ alignas(16) _Float16 lsA[32 * LROW_];

    int tid  = threadIdx.x;
    int lane = tid & 31;
    int wave = tid >> 5;
    int j    = nt * 128 + wave * 16 + (lane & 15);          // gate/up column
    const float* wrow_g = w1 + ((size_t)e * (2 * I_) + j) * H_;
    const float* wrow_u = w1 + ((size_t)e * (2 * I_) + I_ + j) * H_;
    int khalf = (lane < 16) ? 0 : 16;                       // B: K-run select
    int arow  = lane & 15;
    int klo   = (lane < 16) ? 0 : 16;                       // A: K-run select

    // staging assignment: 8 threads per row, contiguous 16B units
    int srow = tid >> 3;
    int tsub = tid & 7;
    int p = mt * 32 + srow;
    int q = (p < cnt_e) ? p : (cnt_e - 1);
    int tok = list[e * T_ + q] >> 1;
    const _Float16* srcrow = xh + (size_t)tok * H_;

    v8f cg0 = {}, cg1 = {}, cu0 = {}, cu1 = {};

    for (int kc = 0; kc < H_; kc += KC_) {
        __syncthreads();
        for (int u = tsub; u < KC_ / 8; u += 8)
            *(v8h*)&lsA[srow * LROW_ + u * 8] = *(const v8h*)&srcrow[kc + u * 8];
        __syncthreads();

        for (int kl = 0; kl < KC_; kl += 32) {
            int k0 = kc + kl;
            v16h bg, bu;
            const float4* pg = (const float4*)(wrow_g + k0 + khalf);
            const float4* pu = (const float4*)(wrow_u + k0 + khalf);
#pragma unroll
            for (int qd = 0; qd < 4; ++qd) {
                float4 fg = pg[qd]; float4 fu = pu[qd];
                bg[4*qd+0] = (_Float16)fg.x; bg[4*qd+1] = (_Float16)fg.y;
                bg[4*qd+2] = (_Float16)fg.z; bg[4*qd+3] = (_Float16)fg.w;
                bu[4*qd+0] = (_Float16)fu.x; bu[4*qd+1] = (_Float16)fu.y;
                bu[4*qd+2] = (_Float16)fu.z; bu[4*qd+3] = (_Float16)fu.w;
            }
            v8h a0l = *(const v8h*)&lsA[arow * LROW_ + kl + klo];
            v8h a0h = *(const v8h*)&lsA[arow * LROW_ + kl + klo + 8];
            v8h a1l = *(const v8h*)&lsA[(16 + arow) * LROW_ + kl + klo];
            v8h a1h = *(const v8h*)&lsA[(16 + arow) * LROW_ + kl + klo + 8];
            v16h a0 = __builtin_shufflevector(a0l, a0h, 0,1,2,3,4,5,6,7,8,9,10,11,12,13,14,15);
            v16h a1 = __builtin_shufflevector(a1l, a1h, 0,1,2,3,4,5,6,7,8,9,10,11,12,13,14,15);
            cg0 = __builtin_amdgcn_wmma_f32_16x16x32_f16(false, a0, false, bg, (short)0, cg0, false, false);
            cu0 = __builtin_amdgcn_wmma_f32_16x16x32_f16(false, a0, false, bu, (short)0, cu0, false, false);
            cg1 = __builtin_amdgcn_wmma_f32_16x16x32_f16(false, a1, false, bg, (short)0, cg1, false, false);
            cu1 = __builtin_amdgcn_wmma_f32_16x16x32_f16(false, a1, false, bu, (short)0, cu1, false, false);
        }
    }

    // epilogue: silu(g)*u -> f16 act (rows >= cnt hold garbage but are never combined)
    int rbase = (lane < 16) ? 0 : 8;
#pragma unroll
    for (int rr = 0; rr < 8; ++rr) {
        int row0 = rr + rbase;
        float g0 = cg0[rr], u0 = cu0[rr];
        float v0 = (g0 / (1.f + __expf(-g0))) * u0;
        act[((size_t)(e * T_ + mt * 32 + row0)) * I_ + j] = (_Float16)v0;
        int row1 = 16 + row0;
        float g1 = cg1[rr], u1 = cu1[rr];
        float v1 = (g1 / (1.f + __expf(-g1))) * u1;
        act[((size_t)(e * T_ + mt * 32 + row1)) * I_ + j] = (_Float16)v1;
    }
}

// ------------------------------------------------------------------
// Kernel 4: GEMM2 (act @ w2^T), scaled scatter into out_k[(t,k)]
// ------------------------------------------------------------------
__global__ __launch_bounds__(256) void moe_gemm2(const float* __restrict__ w2,
                                                 const _Float16* __restrict__ act,
                                                 const int* __restrict__ cnt,
                                                 const int* __restrict__ list,
                                                 const float* __restrict__ wl,
                                                 float* __restrict__ outk) {
    constexpr int MT = T_ / 32;    // 16
    constexpr int NT = H_ / 128;   // 16
    int bx = blockIdx.x;
    int e  = bx / (MT * NT);
    int rm = bx % (MT * NT);
    int mt = rm / NT;
    int nt = rm % NT;
    int cnt_e = cnt[e];
    if (mt * 32 >= cnt_e) return;

    __shared__ alignas(16) _Float16 lsA[32 * LROW_];

    int tid  = threadIdx.x;
    int lane = tid & 31;
    int wave = tid >> 5;
    int hcol = nt * 128 + wave * 16 + (lane & 15);
    const float* wrow = w2 + ((size_t)e * H_ + hcol) * I_;
    int khalf = (lane < 16) ? 0 : 16;
    int arow  = lane & 15;
    int klo   = (lane < 16) ? 0 : 16;

    int srow = tid >> 3;
    int tsub = tid & 7;
    const _Float16* srcrow = act + (size_t)(e * T_ + mt * 32 + srow) * I_;

    v8f c0 = {}, c1 = {};

    for (int kc = 0; kc < I_; kc += KC_) {
        int len = (I_ - kc) < KC_ ? (I_ - kc) : KC_;       // 512,512,384
        __syncthreads();
        for (int u = tsub; u < len / 8; u += 8)
            *(v8h*)&lsA[srow * LROW_ + u * 8] = *(const v8h*)&srcrow[kc + u * 8];
        __syncthreads();

        for (int kl = 0; kl < len; kl += 32) {
            int k0 = kc + kl;
            v16h b;
            const float4* pw = (const float4*)(wrow + k0 + khalf);
#pragma unroll
            for (int qd = 0; qd < 4; ++qd) {
                float4 f = pw[qd];
                b[4*qd+0] = (_Float16)f.x; b[4*qd+1] = (_Float16)f.y;
                b[4*qd+2] = (_Float16)f.z; b[4*qd+3] = (_Float16)f.w;
            }
            v8h a0l = *(const v8h*)&lsA[arow * LROW_ + kl + klo];
            v8h a0h = *(const v8h*)&lsA[arow * LROW_ + kl + klo + 8];
            v8h a1l = *(const v8h*)&lsA[(16 + arow) * LROW_ + kl + klo];
            v8h a1h = *(const v8h*)&lsA[(16 + arow) * LROW_ + kl + klo + 8];
            v16h a0 = __builtin_shufflevector(a0l, a0h, 0,1,2,3,4,5,6,7,8,9,10,11,12,13,14,15);
            v16h a1 = __builtin_shufflevector(a1l, a1h, 0,1,2,3,4,5,6,7,8,9,10,11,12,13,14,15);
            c0 = __builtin_amdgcn_wmma_f32_16x16x32_f16(false, a0, false, b, (short)0, c0, false, false);
            c1 = __builtin_amdgcn_wmma_f32_16x16x32_f16(false, a1, false, b, (short)0, c1, false, false);
        }
    }

    int rbase = (lane < 16) ? 0 : 8;
#pragma unroll
    for (int rr = 0; rr < 8; ++rr) {
        int row0 = rr + rbase;
        int p0 = mt * 32 + row0;
        if (p0 < cnt_e) {
            int pair = list[e * T_ + p0];
            float w  = wl  [e * T_ + p0];
            outk[(size_t)pair * H_ + hcol] = w * c0[rr];
        }
        int p1 = p0 + 16;
        if (p1 < cnt_e) {
            int pair = list[e * T_ + p1];
            float w  = wl  [e * T_ + p1];
            outk[(size_t)pair * H_ + hcol] = w * c1[rr];
        }
    }
}

// ------------------------------------------------------------------
// Kernel 5: combine the K=2 expert contributions per token
// ------------------------------------------------------------------
__global__ __launch_bounds__(256) void moe_combine(const float* __restrict__ outk,
                                                   float* __restrict__ out) {
    size_t idx = ((size_t)blockIdx.x * 256 + threadIdx.x) * 4;  // 1024 blocks -> T*H
    int t = (int)(idx / H_);
    int h = (int)(idx % H_);
    float4 a = *(const float4*)(outk + (size_t)(2 * t)     * H_ + h);
    float4 b = *(const float4*)(outk + (size_t)(2 * t + 1) * H_ + h);
    float4 o;
    o.x = a.x + b.x; o.y = a.y + b.y; o.z = a.z + b.z; o.w = a.w + b.w;
    *(float4*)(out + idx) = o;
}

// ------------------------------------------------------------------
extern "C" void kernel_launch(void* const* d_in, const int* in_sizes, int n_in,
                              void* d_out, int out_size, void* d_ws, size_t ws_size,
                              hipStream_t stream) {
    const float* x  = (const float*)d_in[0];
    const float* w1 = (const float*)d_in[1];
    const float* w2 = (const float*)d_in[2];
    const float* wg = (const float*)d_in[3];
    float* out = (float*)d_out;

    char* ws = (char*)d_ws;
    int*      cnt  = (int*)(ws + OFF_CNT);
    int*      list = (int*)(ws + OFF_LIST);
    float*    wl   = (float*)(ws + OFF_WL);
    _Float16* xh   = (_Float16*)(ws + OFF_XH);
    _Float16* act  = (_Float16*)(ws + OFF_ACT);
    float*    outk = (float*)(ws + OFF_OUTK);

    moe_convert<<<512, 256, 0, stream>>>(x, xh, cnt);
    moe_router <<<T_ / 8, 256, 0, stream>>>(x, wg, cnt, list, wl);
    moe_gemm1  <<<E_ * (T_ / 32) * (I_ / 128), 256, 0, stream>>>(w1, xh, cnt, list, act);
    moe_gemm2  <<<E_ * (T_ / 32) * (H_ / 128), 256, 0, stream>>>(w2, act, cnt, list, wl, outk);
    moe_combine<<<(T_ * H_) / (256 * 4), 256, 0, stream>>>(outk, out);
}
